// TransformerLayer_4088808865866
// MI455X (gfx1250) — matrix operand order repaired and checked by reading, not compiled, not run on hardware
//
#include <hip/hip_runtime.h>

typedef _Float16 f16;
typedef __attribute__((ext_vector_type(16))) _Float16 v16h;
typedef __attribute__((ext_vector_type(8)))  _Float16 v8h;
typedef __attribute__((ext_vector_type(8)))  float    v8f;

#define WMMA_F16(a, b, c) \
  __builtin_amdgcn_wmma_f32_16x16x32_f16(false, (a), false, (b), (short)0, (c), false, false)

// ---------------------------------------------------------------------------
// Fragment loaders (wave32 WMMA; this file's own contiguous k order, same for A and B)
// A 16x32 f16: lane l -> row (l&15); halves 0..15 = K[k0+kb .. +15]
//              contiguous, kb = (l>>4)*16.
// B 32x16 f16 (stored transposed [N,K]): lane l -> col (l&15);
//              halves = K[k0+(l>>4)*16 .. +15] contiguous.
// D 16x16 f32: lane l -> col (l&15); element r -> row (l>>4)*8 + r.
// ---------------------------------------------------------------------------
__device__ __forceinline__ v16h load_afrag(const f16* A, int ldk, int row, int k0, int lane) {
  int kb = (lane >> 4) * 16;
  const f16* p = A + (size_t)row * ldk + k0 + kb;
  union { v16h v; v8h h[2]; } u;
  u.h[0] = *(const v8h*)(p);
  u.h[1] = *(const v8h*)(p + 8);
  return u.v;
}
__device__ __forceinline__ v16h load_bfrag(const f16* Bt, int ldk, int col, int k0, int lane) {
  int kb = (lane >> 4) * 16;
  return *(const v16h*)(Bt + (size_t)col * ldk + k0 + kb);
}
// B fragment out of an LDS-staged chunk: row = col, 32 valid halves, padded pitch
__device__ __forceinline__ v16h load_bfrag_lds(const f16* chunk, int pitch, int col, int lane) {
  const f16* p = chunk + (size_t)col * pitch + (lane >> 4) * 16;
  union { v16h v; v8h h[2]; } u;
  u.h[0] = *(const v8h*)(p);
  u.h[1] = *(const v8h*)(p + 8);
  return u.v;
}

// ---------------------------------------------------------------------------
// starts[b] = lower_bound(batch, b), starts[B] = N   (batch is sorted)
// ---------------------------------------------------------------------------
__global__ void k_starts(const int* __restrict__ batch, int* __restrict__ starts, int Nn, int Bg) {
  int b = threadIdx.x;
  if (b > Bg) return;
  int lo = 0, hi = Nn;
  while (lo < hi) { int mid = (lo + hi) >> 1; if (batch[mid] < b) lo = mid + 1; else hi = mid; }
  starts[b] = lo;
}

__global__ void k_zero16(f16* __restrict__ p, size_t n) {
  size_t i = (size_t)blockIdx.x * blockDim.x + threadIdx.x;
  if (i < n) p[i] = (f16)0.0f;
}

// scatter flat fp32 nodes into dense f16 [B*512, 512]
__global__ void k_scatter(const float* __restrict__ h, const int* __restrict__ batch,
                          const int* __restrict__ starts, f16* __restrict__ hdense) {
  int i = blockIdx.x;
  int b = batch[i];
  int pos = i - starts[b];
  if (pos >= 512) return;
  const float* src = h + (size_t)i * 512;
  f16* dst = hdense + ((size_t)(b * 512 + pos)) * 512;
  for (int d = threadIdx.x; d < 512; d += blockDim.x) dst[d] = (f16)src[d];
}

// convert fp32 W[K,N] -> f16 Wt[N,K]
__global__ void k_wt(const float* __restrict__ W, f16* __restrict__ Wt, int K, int N) {
  size_t idx = (size_t)blockIdx.x * blockDim.x + threadIdx.x;
  if (idx >= (size_t)K * N) return;
  int k = (int)(idx / N), n = (int)(idx % N);
  Wt[(size_t)n * K + k] = (f16)W[idx];
}

// ---------------------------------------------------------------------------
// WMMA GEMM: C[M,Ncols] = A[M,K](f16) @ Wt[Ncols,K](f16)^T + bias
// Block: 8 waves, tile 64x256. Each wave: 32x64 (8 WMMAs / K-step).
// Weight chunk (256 cols x 32 K) staged to LDS via CDNA5 async-to-LDS DMA,
// double-buffered (ASYNCcnt). A fragments register double-buffered. All 4 B
// fragments are pulled from LDS before the 8-WMMA burst (single dscnt wait).
// modes: 0=f16 row-major, 1=f16 [B,H,N,HD] (Q/K), 2=f16 [B,H,HD,N] (V),
//        3=f32 row-major, 4=relu f16
// ---------------------------------------------------------------------------
#define BPITCH 40   // padded halves per staged row (80 B: 16B-aligned, bank-spread)
__global__ void k_gemm(const f16* __restrict__ A, const f16* __restrict__ Wt,
                       const float* __restrict__ bias, void* __restrict__ outp,
                       int M, int Ncols, int K, int mode) {
  __shared__ f16 Bs[2][256 * BPITCH];
  int tid = threadIdx.x;
  int lane = tid & 31;
  int wave = tid >> 5;                      // 0..7
  int tileM  = (blockIdx.y * 2 + (wave >> 2)) * 32;
  int tileN0 = blockIdx.x * 256;            // block N base
  int waveN  = (wave & 3) * 64;             // wave N offset within block

  // async-stage Wt[tileN0+tid][k0..k0+31] (64 B) into Bs[buf][tid*BPITCH]
  auto stage = [&](int buf, int k0) {
    const f16* src = Wt + (size_t)(tileN0 + tid) * K + k0;
    unsigned ldsoff = (unsigned)(size_t)(&Bs[buf][tid * BPITCH]);
    asm volatile("global_load_async_to_lds_b128 %0, %1, off\n\t"
                 "global_load_async_to_lds_b128 %0, %1, off offset:16\n\t"
                 "global_load_async_to_lds_b128 %0, %1, off offset:32\n\t"
                 "global_load_async_to_lds_b128 %0, %1, off offset:48"
                 :: "v"(ldsoff), "v"(src) : "memory");
  };
  auto wait_async = [&] { asm volatile("s_wait_asynccnt 0x0" ::: "memory"); };

  int arow0 = tileM + (lane & 15);
  int arow1 = arow0 + 16;

  stage(0, 0);
  v16h a0 = load_afrag(A, K, arow0, 0, lane);
  v16h a1 = load_afrag(A, K, arow1, 0, lane);
  wait_async();
  __syncthreads();

  v8f acc[2][4] = {};
  int buf = 0;
  for (int k0 = 0; k0 < K; k0 += 32) {
    int kn = k0 + 32;
    bool more = (kn < K);
    if (more) stage(buf ^ 1, kn);
    v16h a0n = a0, a1n = a1;
    if (more) {
      a0n = load_afrag(A, K, arow0, kn, lane);
      a1n = load_afrag(A, K, arow1, kn, lane);
    }
    const f16* chunk = &Bs[buf][0];
    // pull all four B fragments first -> one dscnt wait, then 8-WMMA burst
    v16h bfr[4];
#pragma unroll
    for (int t = 0; t < 4; ++t)
      bfr[t] = load_bfrag_lds(chunk, BPITCH, waveN + t * 16 + (lane & 15), lane);
#pragma unroll
    for (int t = 0; t < 4; ++t) {
      acc[0][t] = WMMA_F16(a0, bfr[t], acc[0][t]);
      acc[1][t] = WMMA_F16(a1, bfr[t], acc[1][t]);
    }
    if (more) wait_async();
    __syncthreads();
    buf ^= 1;
    a0 = a0n; a1 = a1n;
  }

  int nlane = lane & 15;
  int mb = (lane >> 4) * 8;
#pragma unroll
  for (int rt = 0; rt < 2; ++rt) {
#pragma unroll
    for (int t = 0; t < 4; ++t) {
      int n = tileN0 + waveN + t * 16 + nlane;
      float bv = bias[n];
#pragma unroll
      for (int r = 0; r < 8; ++r) {
        int m = tileM + rt * 16 + mb + r;
        float val = acc[rt][t][r] + bv;
        if (mode == 0) {
          ((f16*)outp)[(size_t)m * Ncols + n] = (f16)val;
        } else if (mode == 1) {          // [B,H,NMAX,HD]; col n -> h=n&7, f=n>>3
          int b_ = m >> 9, pos = m & 511, h_ = n & 7, f_ = n >> 3;
          ((f16*)outp)[(((size_t)b_ * 8 + h_) * 512 + pos) * 64 + f_] = (f16)val;
        } else if (mode == 2) {          // [B,H,HD,NMAX]
          int b_ = m >> 9, pos = m & 511, h_ = n & 7, f_ = n >> 3;
          ((f16*)outp)[(((size_t)b_ * 8 + h_) * 64 + f_) * 512 + pos] = (f16)val;
        } else if (mode == 3) {
          ((float*)outp)[(size_t)m * Ncols + n] = val;
        } else {                         // relu f16
          ((f16*)outp)[(size_t)m * Ncols + n] = (f16)(val > 0.f ? val : 0.f);
        }
      }
    }
  }
}

// ---------------------------------------------------------------------------
// Attention: per (b,h): S = Kh(512x64) @ Qh^T, mask cols m>=cnt, softmax over m,
// O = P @ Vh. Each wave owns a 16-row n-tile; S rows in LDS (f32), P in LDS
// (f16). 2 waves/block -> 96 KB dynamic LDS. K fragments hoisted; Q and P@V
// operands register double-buffered.
// ---------------------------------------------------------------------------
#define ATT_WAVES 2
__global__ void k_attn(const f16* __restrict__ qh, const f16* __restrict__ kh,
                       const f16* __restrict__ vhT, f16* __restrict__ av,
                       const int* __restrict__ starts) {
  extern __shared__ char smem[];
  int lane = threadIdx.x & 31;
  int wave = threadIdx.x >> 5;
  float* Srow = (float*)smem + (size_t)wave * 16 * 512;
  f16*   Prow = (f16*)(smem + (size_t)ATT_WAVES * 16 * 512 * 4) + (size_t)wave * 16 * 512;

  int bh = blockIdx.x;                 // b*8 + h
  int b  = bh >> 3, h = bh & 7;
  int n0 = (blockIdx.y * ATT_WAVES + wave) * 16;
  const f16* Kh = kh  + (size_t)bh * 512 * 64;
  const f16* Qh = qh  + (size_t)bh * 512 * 64;
  const f16* VT = vhT + (size_t)bh * 64 * 512;
  int cnt = starts[b + 1] - starts[b];
  if (cnt > 512) cnt = 512;

  const float scale = 0.125f;          // 1/sqrt(HD=64)
  int mlane = lane & 15;
  int nbase = (lane >> 4) * 8;

  // ---- scores: K fragments are invariant across the m-loop; Q double-buffered
  v16h ak0 = load_afrag(Kh, 64, n0 + mlane, 0, lane);
  v16h ak1 = load_afrag(Kh, 64, n0 + mlane, 32, lane);
  v16h q0 = load_bfrag(Qh, 64, mlane, 0, lane);
  v16h q1 = load_bfrag(Qh, 64, mlane, 32, lane);
  for (int m0 = 0; m0 < 512; m0 += 16) {
    v16h q0n = q0, q1n = q1;
    if (m0 + 16 < 512) {
      q0n = load_bfrag(Qh, 64, m0 + 16 + mlane, 0, lane);
      q1n = load_bfrag(Qh, 64, m0 + 16 + mlane, 32, lane);
    }
    v8f s = {};
    s = WMMA_F16(ak0, q0, s);
    s = WMMA_F16(ak1, q1, s);
    int m = m0 + mlane;
    float pen = (m < cnt) ? 0.f : 1e10f;
#pragma unroll
    for (int r = 0; r < 8; ++r)
      Srow[(size_t)(nbase + r) * 512 + m] = s[r] * scale - pen;
    q0 = q0n; q1 = q1n;
  }

  // ---- softmax over m for each of the 16 rows (wave-parallel) ----
  for (int r = 0; r < 16; ++r) {
    float* row = Srow + (size_t)r * 512;
    float mx = -3.0e38f;
    for (int j = lane; j < 512; j += 32) mx = fmaxf(mx, row[j]);
#pragma unroll
    for (int off = 16; off; off >>= 1) mx = fmaxf(mx, __shfl_xor(mx, off, 32));
    float sum = 0.f;
    for (int j = lane; j < 512; j += 32) { float e = __expf(row[j] - mx); sum += e; row[j] = e; }
#pragma unroll
    for (int off = 16; off; off >>= 1) sum += __shfl_xor(sum, off, 32);
    float inv = 1.f / sum;
    f16* prow = Prow + (size_t)r * 512;
    for (int j = lane; j < 512; j += 32) prow[j] = (f16)(row[j] * inv);
  }

  // ---- O(16x64) = P(16x512) @ V(512x64); both operands double-buffered ----
  v8f o[4] = {};
  v16h ap = load_afrag(Prow, 512, mlane, 0, lane);
  v16h bc[4];
#pragma unroll
  for (int t = 0; t < 4; ++t) bc[t] = load_bfrag(VT, 512, t * 16 + mlane, 0, lane);
  for (int k0 = 0; k0 < 512; k0 += 32) {
    int kn = k0 + 32;
    v16h apn = ap;
    v16h bn[4];
#pragma unroll
    for (int t = 0; t < 4; ++t) bn[t] = bc[t];
    if (kn < 512) {
      apn = load_afrag(Prow, 512, mlane, kn, lane);
#pragma unroll
      for (int t = 0; t < 4; ++t) bn[t] = load_bfrag(VT, 512, t * 16 + mlane, kn, lane);
    }
#pragma unroll
    for (int t = 0; t < 4; ++t) o[t] = WMMA_F16(ap, bc[t], o[t]);
    ap = apn;
#pragma unroll
    for (int t = 0; t < 4; ++t) bc[t] = bn[t];
  }
  // scatter back to dense av [8192,512]; feature column = f*8 + h
#pragma unroll
  for (int t = 0; t < 4; ++t) {
    int f = t * 16 + mlane;
#pragma unroll
    for (int r = 0; r < 8; ++r) {
      int n = n0 + nbase + r;
      av[((size_t)(b * 512 + n)) * 512 + f * 8 + h] = (f16)o[t][r];
    }
  }
}

// gather dense projection back to flat nodes + residual
__global__ void k_gather_add(const float* __restrict__ h, const float* __restrict__ dense,
                             const int* __restrict__ batch, const int* __restrict__ starts,
                             float* __restrict__ z) {
  int i = blockIdx.x;
  int b = batch[i];
  int pos = i - starts[b];
  for (int d = threadIdx.x; d < 512; d += blockDim.x) {
    float add = (pos < 512) ? dense[((size_t)(b * 512 + pos)) * 512 + d] : 0.f;
    z[(size_t)i * 512 + d] = h[(size_t)i * 512 + d] + add;
  }
}

__global__ void k_bnstats(const float* __restrict__ x, float* __restrict__ mean,
                          float* __restrict__ rstd, int rows, int cols) {
  __shared__ float bs[256], bs2[256];
  int c = blockIdx.x;
  float s = 0.f, s2 = 0.f;
  for (int i = threadIdx.x; i < rows; i += blockDim.x) {
    float v = x[(size_t)i * cols + c];
    s += v; s2 += v * v;
  }
  bs[threadIdx.x] = s; bs2[threadIdx.x] = s2;
  __syncthreads();
  for (int off = 128; off; off >>= 1) {
    if ((int)threadIdx.x < off) { bs[threadIdx.x] += bs[threadIdx.x + off]; bs2[threadIdx.x] += bs2[threadIdx.x + off]; }
    __syncthreads();
  }
  if (threadIdx.x == 0) {
    float m = bs[0] / rows;
    float var = bs2[0] / rows - m * m;
    mean[c] = m;
    rstd[c] = rsqrtf(var + 1e-5f);
  }
}

__global__ void k_bnapply(const float* __restrict__ x, const float* __restrict__ mean,
                          const float* __restrict__ rstd, const float* __restrict__ g,
                          const float* __restrict__ be, float* __restrict__ yf32,
                          f16* __restrict__ yf16, int cols, size_t total) {
  size_t idx = (size_t)blockIdx.x * blockDim.x + threadIdx.x;
  if (idx >= total) return;
  int c = (int)(idx % cols);
  float v = (x[idx] - mean[c]) * rstd[c] * g[c] + be[c];
  yf32[idx] = v;
  if (yf16) yf16[idx] = (f16)v;
}

__global__ void k_add(const float* __restrict__ a, const float* __restrict__ b,
                      float* __restrict__ z, size_t n) {
  size_t i = (size_t)blockIdx.x * blockDim.x + threadIdx.x;
  if (i < n) z[i] = a[i] + b[i];
}

// ---------------------------------------------------------------------------
extern "C" void kernel_launch(void* const* d_in, const int* in_sizes, int n_in,
                              void* d_out, int out_size, void* d_ws, size_t ws_size,
                              hipStream_t stream) {
  const int Nn = 6144, D = 512, Bg = 16, NMAX = 512, DFF = 1024;
  const float* h   = (const float*)d_in[0];
  const int* batch = (const int*)d_in[1];
  const float* Wq = (const float*)d_in[2];  const float* bq = (const float*)d_in[3];
  const float* Wk = (const float*)d_in[4];  const float* bk = (const float*)d_in[5];
  const float* Wv = (const float*)d_in[6];  const float* bv = (const float*)d_in[7];
  const float* Wo = (const float*)d_in[8];  const float* bo = (const float*)d_in[9];
  const float* g1 = (const float*)d_in[10]; const float* be1 = (const float*)d_in[11];
  const float* W1 = (const float*)d_in[12]; const float* b1 = (const float*)d_in[13];
  const float* W2 = (const float*)d_in[14]; const float* b2 = (const float*)d_in[15];
  const float* g2 = (const float*)d_in[16]; const float* be2 = (const float*)d_in[17];

  char* w = (char*)d_ws;
  size_t off = 0;
  auto alloc = [&](size_t bytes) -> char* {
    char* p = w + off;
    off += (bytes + 255) & ~(size_t)255;
    return p;
  };
  const size_t MD = (size_t)Bg * NMAX;          // 8192 dense rows
  int*  starts = (int*) alloc(64 * sizeof(int));
  f16*  hdense = (f16*) alloc(MD * D * 2);
  f16*  WqT    = (f16*) alloc((size_t)D * D * 2);
  f16*  WkT    = (f16*) alloc((size_t)D * D * 2);
  f16*  WvT    = (f16*) alloc((size_t)D * D * 2);
  f16*  WoT    = (f16*) alloc((size_t)D * D * 2);
  f16*  W1T    = (f16*) alloc((size_t)DFF * D * 2);   // [1024,512]
  f16*  W2T    = (f16*) alloc((size_t)D * DFF * 2);   // [512,1024]
  f16*  qhb    = (f16*) alloc(MD * D * 2);
  f16*  khb    = (f16*) alloc(MD * D * 2);
  f16*  vhT    = (f16*) alloc(MD * D * 2);
  f16*  av     = (f16*) alloc(MD * D * 2);
  float* dense_o = (float*)alloc(MD * D * 4);
  float* zbuf  = (float*)alloc((size_t)Nn * D * 4);
  float* h1f32 = (float*)alloc((size_t)Nn * D * 4);
  f16*  h1f16  = (f16*) alloc((size_t)Nn * D * 2);
  f16*  ff1    = (f16*) alloc((size_t)Nn * DFF * 2);
  float* ff2   = (float*)alloc((size_t)Nn * D * 4);
  float* mean1 = (float*)alloc(D * 4);
  float* rstd1 = (float*)alloc(D * 4);
  float* mean2 = (float*)alloc(D * 4);
  float* rstd2 = (float*)alloc(D * 4);
  (void)ws_size; (void)in_sizes; (void)n_in; (void)out_size;

  auto gemm = [&](const f16* A, const f16* Wt, const float* bias, void* out,
                  int M, int Ncols, int K, int mode) {
    dim3 grid(Ncols / 256, M / 64);
    k_gemm<<<grid, 256, 0, stream>>>(A, Wt, bias, out, M, Ncols, K, mode);
  };

  // 0) graph offsets
  k_starts<<<1, 32, 0, stream>>>(batch, starts, Nn, Bg);
  // 1) dense scatter (zero-fill then scatter + f32->f16)
  {
    size_t n = MD * D;
    k_zero16<<<(unsigned)((n + 255) / 256), 256, 0, stream>>>(hdense, n);
    k_scatter<<<Nn, 256, 0, stream>>>(h, batch, starts, hdense);
  }
  // 2) weight transpose + f16 convert
  k_wt<<<(D * D + 255) / 256, 256, 0, stream>>>(Wq, WqT, D, D);
  k_wt<<<(D * D + 255) / 256, 256, 0, stream>>>(Wk, WkT, D, D);
  k_wt<<<(D * D + 255) / 256, 256, 0, stream>>>(Wv, WvT, D, D);
  k_wt<<<(D * D + 255) / 256, 256, 0, stream>>>(Wo, WoT, D, D);
  k_wt<<<(D * DFF + 255) / 256, 256, 0, stream>>>(W1, W1T, D, DFF);
  k_wt<<<(DFF * D + 255) / 256, 256, 0, stream>>>(W2, W2T, DFF, D);
  // 3) QKV projections (head-major epilogues)
  gemm(hdense, WqT, bq, qhb, (int)MD, D, D, 1);
  gemm(hdense, WkT, bk, khb, (int)MD, D, D, 1);
  gemm(hdense, WvT, bv, vhT, (int)MD, D, D, 2);
  // 4) attention (128 (b,h) pairs x 16 n-tile groups; 96KB dynamic LDS)
  {
    dim3 grid(Bg * 8, NMAX / (16 * ATT_WAVES));
    size_t shmem = (size_t)ATT_WAVES * 16 * 512 * 4 + (size_t)ATT_WAVES * 16 * 512 * 2;
    k_attn<<<grid, ATT_WAVES * 32, shmem, stream>>>(qhb, khb, vhT, av, starts);
  }
  // 5) output projection (dense, f32)
  gemm(av, WoT, bo, dense_o, (int)MD, D, D, 3);
  // 6) gather + residual, BN1
  k_gather_add<<<Nn, 256, 0, stream>>>(h, dense_o, batch, starts, zbuf);
  k_bnstats<<<D, 256, 0, stream>>>(zbuf, mean1, rstd1, Nn, D);
  {
    size_t total = (size_t)Nn * D;
    k_bnapply<<<(unsigned)((total + 255) / 256), 256, 0, stream>>>(
        zbuf, mean1, rstd1, g1, be1, h1f32, h1f16, D, total);
  }
  // 7) FFN
  gemm(h1f16, W1T, b1, ff1, Nn, DFF, D, 4);   // relu -> f16
  gemm(ff1, W2T, b2, ff2, Nn, D, DFF, 3);     // -> f32
  // 8) residual2 + BN2 -> d_out
  {
    size_t total = (size_t)Nn * D;
    k_add<<<(unsigned)((total + 255) / 256), 256, 0, stream>>>(h1f32, ff2, zbuf, total);
    k_bnstats<<<D, 256, 0, stream>>>(zbuf, mean2, rstd2, Nn, D);
    k_bnapply<<<(unsigned)((total + 255) / 256), 256, 0, stream>>>(
        zbuf, mean2, rstd2, g2, be2, (float*)d_out, (f16*)nullptr, D, total);
  }
}
